// Transformer_84902913508079
// MI455X (gfx1250) — matrix operand order repaired and checked by reading, not compiled, not run on hardware
//
#include <hip/hip_runtime.h>
#include <hip/hip_bf16.h>
#include <cstdint>

typedef __attribute__((ext_vector_type(16))) _Float16 v16h;
typedef __attribute__((ext_vector_type(8)))  _Float16 v8h;
typedef __attribute__((ext_vector_type(4)))  _Float16 v4h;
typedef __attribute__((ext_vector_type(8)))  float    v8f;
typedef __attribute__((ext_vector_type(4)))  float    v4f;

#define DEVI static __device__ __forceinline__

// ---------------- constants ----------------
constexpr int BATCH = 8;
constexpr int NTOK  = 1024;   // 32*32
constexpr int DMODEL= 512;
constexpr int HEADS = 8;
constexpr int DH    = 64;
constexpr int E3    = 3 * DMODEL;       // 1536
constexpr int ROWS  = BATCH * NTOK;     // 8192 flattened rows
constexpr float LN_EPS = 1e-5f;

// ---------------- WMMA helpers ----------------
DEVI v8f wmma_f16(v16h a, v16h b, v8f c) {
  // D = A(16x32 f16) x B(32x16 f16) + C(16x16 f32)
  return __builtin_amdgcn_wmma_f32_16x16x32_f16(
      /*neg_a=*/false, a, /*neg_b=*/false, b,
      /*c_mod=*/(short)0, c, /*reuse_a=*/false, /*reuse_b=*/false);
}

DEVI v16h pack16(v8h lo, v8h hi) {
  v16h r;
#pragma unroll
  for (int i = 0; i < 8; ++i) { r[i] = lo[i]; r[i + 8] = hi[i]; }
  return r;
}

// Load A-fragment (row-major f16 matrix, leading dim lda) tile at (row0,k0).
// Layout used here: lane L<16: M=L, K = sub*16+[0..15] contiguous (sub=lane>>4)
DEVI v16h load_a_frag(const _Float16* __restrict__ A, size_t lda,
                      int row0, int k0, int l, int sub) {
  const _Float16* p = A + (size_t)(row0 + l) * lda + k0 + sub * 16;
  v8h lo = *(const v8h*)(p);
  v8h hi = *(const v8h*)(p + 8);
  return pack16(lo, hi);
}

// Load B-fragment for C = A * W^T. W row-major [N,ldb]; tile cols n0..n0+15,
// K-slab k0..k0+31. Lane L(&15) = column n0+L, K = sub*16 + [0..15] contiguous.
DEVI v16h load_bT_frag(const _Float16* __restrict__ W, size_t ldb,
                       int n0, int k0, int l, int sub) {
  return *(const v16h*)(W + (size_t)(n0 + l) * ldb + k0 + sub * 16);
}

// ---------------- prep: NCHW -> [b,n,d] + sinusoidal PE ----------------
__global__ void k_prep(const float* __restrict__ x, float* __restrict__ t) {
  size_t i = (size_t)blockIdx.x * blockDim.x + threadIdx.x;
  if (i >= (size_t)ROWS * DMODEL) return;
  int c   = (int)(i & (DMODEL - 1));
  size_t bp = i >> 9;                 // /512
  int pos = (int)(bp & (NTOK - 1));
  int b   = (int)(bp >> 10);
  float xi = x[((size_t)b * DMODEL + c) * NTOK + pos];
  int i2 = c & ~1;
  // 10000^(-i2/512) = exp(-i2*ln(10000)/512)
  float inv = __expf(-(float)i2 * (9.2103403719761836f / 512.0f));
  float ang = (float)pos * inv;
  float pe = (c & 1) ? __cosf(ang) : __sinf(ang);
  t[i] = xi + pe;
}

// ---------------- f32 -> f16 convert ----------------
__global__ void k_cvt(const float* __restrict__ s, _Float16* __restrict__ d, int n) {
  int i = blockIdx.x * blockDim.x + threadIdx.x;
  if (i < n) d[i] = (_Float16)s[i];
}

// ---------------- LayerNorm (wave per row of 512) -> f16 ----------------
__global__ void k_ln_f16(const float* __restrict__ t,
                         const float* __restrict__ w,
                         const float* __restrict__ bb,
                         _Float16* __restrict__ out) {
  int wave = (blockIdx.x * blockDim.x + threadIdx.x) >> 5;
  int lane = threadIdx.x & 31;
  if (wave >= ROWS) return;
  const float* row = t + (size_t)wave * DMODEL;
  float vals[16];
  float s = 0.f, s2 = 0.f;
#pragma unroll
  for (int c = 0; c < 4; ++c) {
    v4f v = *(const v4f*)(row + c * 128 + lane * 4);
#pragma unroll
    for (int j = 0; j < 4; ++j) { float f = v[j]; vals[c*4+j] = f; s += f; s2 += f*f; }
  }
#pragma unroll
  for (int m = 1; m < 32; m <<= 1) { s += __shfl_xor(s, m); s2 += __shfl_xor(s2, m); }
  float mean = s * (1.0f / DMODEL);
  float var  = s2 * (1.0f / DMODEL) - mean * mean;
  float rstd = rsqrtf(var + LN_EPS);
  _Float16* orow = out + (size_t)wave * DMODEL;
#pragma unroll
  for (int c = 0; c < 4; ++c) {
    int col = c * 128 + lane * 4;
    v4f wv = *(const v4f*)(w + col);
    v4f bv = *(const v4f*)(bb + col);
    v4h h;
#pragma unroll
    for (int j = 0; j < 4; ++j)
      h[j] = (_Float16)((vals[c*4+j] - mean) * rstd * wv[j] + bv[j]);
    *(v4h*)(orow + col) = h;
  }
}

// ---------------- final LayerNorm -> fp32, transposed to [b,c,pos] ----------------
__global__ void k_ln_out(const float* __restrict__ t,
                         const float* __restrict__ w,
                         const float* __restrict__ bb,
                         float* __restrict__ out) {
  int wave = (blockIdx.x * blockDim.x + threadIdx.x) >> 5;
  int lane = threadIdx.x & 31;
  if (wave >= ROWS) return;
  int b = wave >> 10, pos = wave & (NTOK - 1);
  const float* row = t + (size_t)wave * DMODEL;
  float vals[16];
  float s = 0.f, s2 = 0.f;
#pragma unroll
  for (int c = 0; c < 4; ++c) {
    v4f v = *(const v4f*)(row + c * 128 + lane * 4);
#pragma unroll
    for (int j = 0; j < 4; ++j) { float f = v[j]; vals[c*4+j] = f; s += f; s2 += f*f; }
  }
#pragma unroll
  for (int m = 1; m < 32; m <<= 1) { s += __shfl_xor(s, m); s2 += __shfl_xor(s2, m); }
  float mean = s * (1.0f / DMODEL);
  float var  = s2 * (1.0f / DMODEL) - mean * mean;
  float rstd = rsqrtf(var + LN_EPS);
#pragma unroll
  for (int c = 0; c < 4; ++c) {
    int col = c * 128 + lane * 4;
    v4f wv = *(const v4f*)(w + col);
    v4f bv = *(const v4f*)(bb + col);
#pragma unroll
    for (int j = 0; j < 4; ++j) {
      float v = (vals[c*4+j] - mean) * rstd * wv[j] + bv[j];
      out[((size_t)b * DMODEL + col + j) * NTOK + pos] = v;
    }
  }
}

// ---------------- WMMA GEMM: C[M,N] = A[M,K](f16) * W[N,K]^T (+bias, epilogue) ---
// MODE 0: store f16 (no bias).  MODE 1: +bias, exact GELU, store f16.
// MODE 2: +bias, fp32 residual add into resid[M,N].
// Per wave: 32x64 output tile (2 M-frags x 4 N-frags).
template <int MODE>
__global__ void k_gemm(const _Float16* __restrict__ A,
                       const _Float16* __restrict__ W,
                       const float* __restrict__ bias,
                       _Float16* __restrict__ out16,
                       float* __restrict__ resid,
                       int M, int N, int K) {
  const int wave = (blockIdx.x * blockDim.x + threadIdx.x) >> 5;
  const int lane = threadIdx.x & 31;
  const int ntiles = N >> 6;
  const int m0 = (wave / ntiles) * 32;
  const int n0 = (wave % ntiles) * 64;
  if (m0 >= M) return;
  const int l = lane & 15, sub = lane >> 4;

  v8f c[2][4];
#pragma unroll
  for (int i = 0; i < 2; ++i)
#pragma unroll
    for (int j = 0; j < 4; ++j)
#pragma unroll
      for (int e = 0; e < 8; ++e) c[i][j][e] = 0.0f;

  for (int k0 = 0; k0 < K; k0 += 32) {
    v16h a0 = load_a_frag(A, K, m0,      k0, l, sub);
    v16h a1 = load_a_frag(A, K, m0 + 16, k0, l, sub);
    if (k0 + 32 < K)
      __builtin_prefetch(W + (size_t)(n0 + l) * K + k0 + 32, 0, 1);
#pragma unroll
    for (int j = 0; j < 4; ++j) {
      v16h bf = load_bT_frag(W, K, n0 + j * 16, k0, l, sub);
      c[0][j] = wmma_f16(a0, bf, c[0][j]);
      c[1][j] = wmma_f16(a1, bf, c[1][j]);
    }
  }

#pragma unroll
  for (int mt = 0; mt < 2; ++mt) {
    int rowbase = m0 + mt * 16 + sub * 8;
#pragma unroll
    for (int j = 0; j < 4; ++j) {
      int col = n0 + j * 16 + l;
      float bv = (MODE != 0) ? bias[col] : 0.0f;
#pragma unroll
      for (int r = 0; r < 8; ++r) {
        float v = c[mt][j][r] + bv;
        if constexpr (MODE == 1)
          v = 0.5f * v * (1.0f + erff(v * 0.70710678118654752f));
        if constexpr (MODE == 2)
          resid[(size_t)(rowbase + r) * N + col] += v;
        else
          out16[(size_t)(rowbase + r) * N + col] = (_Float16)v;
      }
    }
  }
}

// ---------------- V transpose: vt[b,h,f,j] = qkv[b,j,1024+h*64+f] --------------
__global__ void k_vt(const _Float16* __restrict__ qkv, _Float16* __restrict__ vt) {
  size_t i = (size_t)blockIdx.x * blockDim.x + threadIdx.x;
  if (i >= (size_t)BATCH * HEADS * DH * NTOK) return;
  int j = (int)(i & (NTOK - 1));
  size_t rest = i >> 10;
  int f = (int)(rest & (DH - 1)); rest >>= 6;
  int h = (int)(rest & (HEADS - 1));
  int b = (int)(rest >> 3);
  vt[i] = qkv[(size_t)b * NTOK * E3 + (size_t)j * E3 + 2 * DMODEL + h * DH + f];
}

// ---------------- flash attention: wave per (b, head, 16-row q block) ----------
__global__ __launch_bounds__(256) void k_attn(const _Float16* __restrict__ qkv,
                                              const _Float16* __restrict__ vt,
                                              _Float16* __restrict__ o16) {
  __shared__ __align__(64) _Float16 plds[8][16 * 32];
  const int wib  = threadIdx.x >> 5;
  const int gwave = blockIdx.x * 8 + wib;
  if (gwave >= BATCH * HEADS * (NTOK / 16)) return;
  const int lane = threadIdx.x & 31;
  const int l = lane & 15, sub = lane >> 4;
  const int qb = gwave & 63;
  const int h  = (gwave >> 6) & (HEADS - 1);
  const int b  = gwave >> 9;

  const size_t bbase = (size_t)b * NTOK * E3;
  const _Float16* qrow = qkv + bbase + (size_t)(qb * 16 + l) * E3 + h * DH;

  v16h qa[2];
#pragma unroll
  for (int ks = 0; ks < 2; ++ks) {
    const _Float16* p = qrow + ks * 32 + sub * 16;
    qa[ks] = pack16(*(const v8h*)(p), *(const v8h*)(p + 8));
  }

  v8f o[4];
  float rm[8], rs[8];
#pragma unroll
  for (int t = 0; t < 4; ++t)
#pragma unroll
    for (int r = 0; r < 8; ++r) o[t][r] = 0.0f;
#pragma unroll
  for (int r = 0; r < 8; ++r) { rm[r] = -3.0e38f; rs[r] = 0.0f; }

  const _Float16* kbase = qkv + bbase + DMODEL + h * DH;
  const _Float16* vbase = vt + (size_t)(b * HEADS + h) * DH * NTOK;
  _Float16* pl = &plds[wib][0];
  const float scale = 0.125f;  // dh^-0.5, dh=64

  for (int kv0 = 0; kv0 < NTOK; kv0 += 32) {
    v8f s0, s1;
#pragma unroll
    for (int r = 0; r < 8; ++r) { s0[r] = 0.0f; s1[r] = 0.0f; }
#pragma unroll
    for (int ks = 0; ks < 2; ++ks) {
      int k0 = ks * 32;
      v16h kb0 = *(const v16h*)(kbase + (size_t)(kv0 + l)      * E3 + k0 + sub * 16);
      v16h kb1 = *(const v16h*)(kbase + (size_t)(kv0 + 16 + l) * E3 + k0 + sub * 16);
      s0 = wmma_f16(qa[ks], kb0, s0);
      s1 = wmma_f16(qa[ks], kb1, s1);
    }
    float alph[8];
#pragma unroll
    for (int r = 0; r < 8; ++r) {
      float a = s0[r] * scale, bq = s1[r] * scale;
      float mx = fmaxf(a, bq);
#pragma unroll
      for (int m = 1; m < 16; m <<= 1) mx = fmaxf(mx, __shfl_xor(mx, m));
      float nm = fmaxf(rm[r], mx);
      float alpha = __expf(rm[r] - nm);
      float p0 = __expf(a - nm), p1 = __expf(bq - nm);
      float ps = p0 + p1;
#pragma unroll
      for (int m = 1; m < 16; m <<= 1) ps += __shfl_xor(ps, m);
      rs[r] = rs[r] * alpha + ps;
      rm[r] = nm;
      alph[r] = alpha;
      pl[(sub * 8 + r) * 32 + l]      = (_Float16)p0;
      pl[(sub * 8 + r) * 32 + 16 + l] = (_Float16)p1;
    }
#pragma unroll
    for (int t = 0; t < 4; ++t)
#pragma unroll
      for (int r = 0; r < 8; ++r) o[t][r] *= alph[r];
    // P (C-layout) -> A-layout via LDS (same-wave DS ops are in-order)
    v16h pa = pack16(*(const v8h*)(pl + l * 32 + sub * 16),
                     *(const v8h*)(pl + l * 32 + 8 + sub * 16));
#pragma unroll
    for (int t = 0; t < 4; ++t) {
      v16h vb = *(const v16h*)(vbase + (size_t)(t * 16 + l) * NTOK + kv0 + sub * 16);
      o[t] = wmma_f16(pa, vb, o[t]);
    }
  }

#pragma unroll
  for (int r = 0; r < 8; ++r) rs[r] = 1.0f / rs[r];
#pragma unroll
  for (int t = 0; t < 4; ++t)
#pragma unroll
    for (int r = 0; r < 8; ++r) {
      size_t row = (size_t)b * NTOK + qb * 16 + sub * 8 + r;
      o16[row * DMODEL + h * DH + t * 16 + l] = (_Float16)(o[t][r] * rs[r]);
    }
}

// ---------------- host-side orchestration ----------------
extern "C" void kernel_launch(void* const* d_in, const int* in_sizes, int n_in,
                              void* d_out, int out_size, void* d_ws, size_t ws_size,
                              hipStream_t stream) {
  const float* x         = (const float*)d_in[0];
  const float* attn_ln_w = (const float*)d_in[1];
  const float* attn_ln_b = (const float*)d_in[2];
  const float* w_qkv     = (const float*)d_in[3];
  const float* w_out     = (const float*)d_in[4];
  const float* b_out     = (const float*)d_in[5];
  const float* ffn_ln_w  = (const float*)d_in[6];
  const float* ffn_ln_b  = (const float*)d_in[7];
  const float* w1        = (const float*)d_in[8];
  const float* b1        = (const float*)d_in[9];
  const float* w2        = (const float*)d_in[10];
  const float* b2        = (const float*)d_in[11];
  const float* ln_w      = (const float*)d_in[12];
  const float* ln_b      = (const float*)d_in[13];

  uint8_t* ws = (uint8_t*)d_ws;
  size_t off = 0;
  float*    t      = (float*)   (ws + off); off += (size_t)ROWS * DMODEL * 4;      // 16 MB
  _Float16* h16    = (_Float16*)(ws + off); off += (size_t)ROWS * DMODEL * 2;      //  8 MB
  _Float16* qkv16  = (_Float16*)(ws + off); off += (size_t)ROWS * E3 * 2;          // 24 MB
  _Float16* vt16   = (_Float16*)(ws + off); off += (size_t)ROWS * DMODEL * 2;      //  8 MB
  _Float16* o16    = (_Float16*)(ws + off); off += (size_t)ROWS * DMODEL * 2;      //  8 MB
  _Float16* g16    = (_Float16*)(ws + off); off += (size_t)ROWS * 2 * DMODEL * 2;  // 16 MB
  _Float16* wqkv16 = (_Float16*)(ws + off); off += (size_t)4 * E3 * DMODEL * 2;
  _Float16* wout16 = (_Float16*)(ws + off); off += (size_t)4 * DMODEL * DMODEL * 2;
  _Float16* w116   = (_Float16*)(ws + off); off += (size_t)2 * DMODEL * DMODEL * 2;
  _Float16* w216   = (_Float16*)(ws + off); off += (size_t)2 * DMODEL * DMODEL * 2;

  const int TPB = 256;
  const size_t nelem = (size_t)ROWS * DMODEL;  // 4,194,304

  k_prep<<<(unsigned)((nelem + TPB - 1) / TPB), TPB, 0, stream>>>(x, t);

  k_cvt<<<(4 * E3 * DMODEL + TPB - 1) / TPB, TPB, 0, stream>>>(w_qkv, wqkv16, 4 * E3 * DMODEL);
  k_cvt<<<(4 * DMODEL * DMODEL + TPB - 1) / TPB, TPB, 0, stream>>>(w_out, wout16, 4 * DMODEL * DMODEL);
  k_cvt<<<(2 * DMODEL * DMODEL + TPB - 1) / TPB, TPB, 0, stream>>>(w1, w116, 2 * DMODEL * DMODEL);
  k_cvt<<<(2 * DMODEL * DMODEL + TPB - 1) / TPB, TPB, 0, stream>>>(w2, w216, 2 * DMODEL * DMODEL);

  const int LN_BLOCKS = ROWS * 32 / TPB;  // 1024

  for (int L = 0; L < 4; ++L) {
    // pre-LN (attention)
    k_ln_f16<<<LN_BLOCKS, TPB, 0, stream>>>(t, attn_ln_w + L * DMODEL,
                                            attn_ln_b + L * DMODEL, h16);
    // qkv = h16 @ Wqkv^T : M=8192 N=1536 K=512 -> 6144 waves
    k_gemm<0><<<(ROWS / 32) * (E3 / 64) / 8, TPB, 0, stream>>>(
        h16, wqkv16 + (size_t)L * E3 * DMODEL, nullptr, qkv16, nullptr,
        ROWS, E3, DMODEL);
    // V transpose
    k_vt<<<(unsigned)((nelem + TPB - 1) / TPB), TPB, 0, stream>>>(qkv16, vt16);
    // flash attention: 4096 waves
    k_attn<<<BATCH * HEADS * (NTOK / 16) / 8, TPB, 0, stream>>>(qkv16, vt16, o16);
    // out projection + bias + residual into t : M=8192 N=512 K=512
    k_gemm<2><<<(ROWS / 32) * (DMODEL / 64) / 8, TPB, 0, stream>>>(
        o16, wout16 + (size_t)L * DMODEL * DMODEL, b_out + L * DMODEL,
        nullptr, t, ROWS, DMODEL, DMODEL);
    // FFN pre-LN
    k_ln_f16<<<LN_BLOCKS, TPB, 0, stream>>>(t, ffn_ln_w, ffn_ln_b, h16);
    // FFN1 + bias + exact GELU : M=8192 N=1024 K=512
    k_gemm<1><<<(ROWS / 32) * (2 * DMODEL / 64) / 8, TPB, 0, stream>>>(
        h16, w116, b1, g16, nullptr, ROWS, 2 * DMODEL, DMODEL);
    // FFN2 + bias + residual into t : M=8192 N=512 K=1024
    k_gemm<2><<<(ROWS / 32) * (DMODEL / 64) / 8, TPB, 0, stream>>>(
        g16, w216, b2, nullptr, t, ROWS, DMODEL, 2 * DMODEL);
  }

  // final LN, transpose back to [b, c, 32*32]
  k_ln_out<<<LN_BLOCKS, TPB, 0, stream>>>(t, ln_w, ln_b, (float*)d_out);
}
